// CascadedGruCell_60825326846602
// MI455X (gfx1250) — matrix operand order repaired and checked by reading, not compiled, not run on hardware
//
#include <hip/hip_runtime.h>
#include <cstdint>

#define B_ROWS 65536
#define F_DIM  512
#define V_DIM  28
#define NCOLS  112           // 3V + V = 112 = 7*16
#define NTILES 7
#define KSTEPS 16            // 512 / 32
#define ROWS_PER_BLOCK 64
#define THREADS 64           // 2 waves (wave32), 32 rows per wave

typedef __attribute__((ext_vector_type(16))) __bf16 v16bf;
typedef __attribute__((ext_vector_type(8)))  float  v8f;

struct alignas(16) Frag  { uint32_t w[8]; };  // raw 16 bf16 (for packed global loads)
struct alignas(16) BFrag { __bf16  h[16]; };  // typed 16 bf16 (hardware v_cvt_pk_bf16_f32)

__device__ __forceinline__ uint32_t f2bf(float f) {
  uint32_t u = __float_as_uint(f);
  return (u + 0x7FFFu + ((u >> 16) & 1u)) >> 16;   // round-to-nearest-even
}

#define WMMA_BF16(A, Bm, C) \
  __builtin_amdgcn_wmma_f32_16x16x32_bf16(false, (A), false, (Bm), (short)0, (C), false, false)

// ---------------------------------------------------------------------------
// Prep: pack W_big = [gru_kernel | Co] (512x112) and W_small = [rkernel | Uo]
// (28x112, K padded to 32) into per-lane WMMA B-fragment order, bf16.
// B layout (16x16x32 bf16): lane n (0..15) holds col N=n, K=0..7,16..23
// (VGPR j<4 = K 2j,2j+1; j>=4 = K 8+2j,9+2j); lanes 16..31 hold same cols, K=8..15,24..31.
// Also build table[k] = emb[k,:] . Wo  (the WoY gather collapses to a LUT).
// ---------------------------------------------------------------------------
__global__ void cgru_prep(const float* __restrict__ gk, const float* __restrict__ grk,
                          const float* __restrict__ Wo, const float* __restrict__ Uo,
                          const float* __restrict__ Co, const float* __restrict__ emb,
                          uint16_t* __restrict__ wbig, uint16_t* __restrict__ wsmall,
                          float* __restrict__ table) {
  const int tid = blockIdx.x * blockDim.x + threadIdx.x;
  const int nth = gridDim.x * blockDim.x;

  if (blockIdx.x == 0 && threadIdx.x < V_DIM) {
    float s = 0.f;
    for (int e = 0; e < V_DIM; ++e) s += emb[threadIdx.x * V_DIM + e] * Wo[e];
    table[threadIdx.x] = s;
  }

  const int NB = NTILES * KSTEPS * 32 * 16;
  for (int e = tid; e < NB; e += nth) {
    int q = e & 15, l = (e >> 4) & 31, ks = (e >> 9) & 15, nt = e >> 13;
    int k = ks * 32 + (q & 7) + ((l >= 16) ? 8 : 0) + ((q >= 8) ? 16 : 0);
    int n = nt * 16 + (l & 15);
    float v = (n < 84) ? gk[k * 84 + n] : Co[k * 28 + (n - 84)];
    wbig[e] = (uint16_t)f2bf(v);
  }

  const int NS = NTILES * 32 * 16;
  for (int e = tid; e < NS; e += nth) {
    int q = e & 15, l = (e >> 4) & 31, nt = e >> 9;
    int k = (q & 7) + ((l >= 16) ? 8 : 0) + ((q >= 8) ? 16 : 0);
    int n = nt * 16 + (l & 15);
    float v = 0.f;
    if (k < V_DIM) v = (n < 84) ? grk[k * 84 + n] : Uo[k * 28 + (n - 84)];
    wsmall[e] = (uint16_t)f2bf(v);
  }
}

__device__ __forceinline__ v16bf make_a_frag(const float* arow, int kb, int o) {
  const float4* p1 = reinterpret_cast<const float4*>(arow + kb + o);
  const float4* p2 = reinterpret_cast<const float4*>(arow + kb + 16 + o);
  float4 a0 = p1[0], a1 = p1[1];
  float4 b0 = p2[0], b1 = p2[1];
  BFrag af;
  af.h[0]  = (__bf16)a0.x; af.h[1]  = (__bf16)a0.y;
  af.h[2]  = (__bf16)a0.z; af.h[3]  = (__bf16)a0.w;
  af.h[4]  = (__bf16)a1.x; af.h[5]  = (__bf16)a1.y;
  af.h[6]  = (__bf16)a1.z; af.h[7]  = (__bf16)a1.w;
  af.h[8]  = (__bf16)b0.x; af.h[9]  = (__bf16)b0.y;
  af.h[10] = (__bf16)b0.z; af.h[11] = (__bf16)b0.w;
  af.h[12] = (__bf16)b1.x; af.h[13] = (__bf16)b1.y;
  af.h[14] = (__bf16)b1.z; af.h[15] = (__bf16)b1.w;
  return __builtin_bit_cast(v16bf, af);
}

// ---------------------------------------------------------------------------
// Fused main kernel: per 64-row block (2 waves, 32 rows each)
//   big:   [64 x 512] x [512 x 112]  (xm | CoC)   bf16 WMMA, f32 accum
//   small: [64 x 32]  x [32 x 112]   (hm | UoH)
//   then GRU gates + logits + softmax, one thread per row (all 64 active).
// Each B fragment feeds two WMMAs (M=32 per wave) to amortize B loads.
// ---------------------------------------------------------------------------
__global__ void __launch_bounds__(THREADS) cgru_fused(
    const float* __restrict__ inputs, const int* __restrict__ prev_pred,
    const float* __restrict__ prev_state, const float* __restrict__ gbias,
    const float* __restrict__ Bo, const uint16_t* __restrict__ wbig,
    const uint16_t* __restrict__ wsmall, const float* __restrict__ table,
    float* __restrict__ out)
{
  __shared__ float ldsBig[ROWS_PER_BLOCK * NCOLS];
  __shared__ float ldsSmall[ROWS_PER_BLOCK * NCOLS];

  const int tid  = threadIdx.x;
  const int wave = tid >> 5;              // 0..1
  const int lane = tid & 31;
  const int l15  = lane & 15;
  const int hsel = (lane < 16) ? 0 : 1;
  const int rowLocal = wave * 32;         // 32-row strip per wave
  const long rowLo = (long)blockIdx.x * ROWS_PER_BLOCK + rowLocal + l15;
  const float* arow0 = inputs + rowLo * F_DIM;            // rows M=0..15 of strip
  const float* arow1 = arow0 + 16 * F_DIM;                // rows M=16..31 of strip
  const int o = hsel ? 8 : 0;             // A-layout K offset for this lane group

  const Frag* bigFrags   = reinterpret_cast<const Frag*>(wbig);
  const Frag* smallFrags = reinterpret_cast<const Frag*>(wsmall);

  v8f zero = {};
  v8f aL0 = zero, aL1 = zero, aL2 = zero, aL3 = zero, aL4 = zero, aL5 = zero, aL6 = zero;
  v8f aH0 = zero, aH1 = zero, aH2 = zero, aH3 = zero, aH4 = zero, aH5 = zero, aH6 = zero;

  // ---- big GEMM k-loop: 16 steps of K=32, 14 WMMAs per step ----
  for (int ks = 0; ks < KSTEPS; ++ks) {
    const int kb = ks * 32;
    if (ks + 1 < KSTEPS) {
      __builtin_prefetch(arow0 + kb + 32, 0, 1);
      __builtin_prefetch(arow1 + kb + 32, 0, 1);
    }
    v16bf A0 = make_a_frag(arow0, kb, o);
    v16bf A1 = make_a_frag(arow1, kb, o);

    const Frag* fb = bigFrags + (size_t)ks * 32 + lane;
    {
      v16bf Bm = __builtin_bit_cast(v16bf, fb[0 * KSTEPS * 32]);
      aL0 = WMMA_BF16(A0, Bm, aL0);  aH0 = WMMA_BF16(A1, Bm, aH0);
    }
    {
      v16bf Bm = __builtin_bit_cast(v16bf, fb[1 * KSTEPS * 32]);
      aL1 = WMMA_BF16(A0, Bm, aL1);  aH1 = WMMA_BF16(A1, Bm, aH1);
    }
    {
      v16bf Bm = __builtin_bit_cast(v16bf, fb[2 * KSTEPS * 32]);
      aL2 = WMMA_BF16(A0, Bm, aL2);  aH2 = WMMA_BF16(A1, Bm, aH2);
    }
    {
      v16bf Bm = __builtin_bit_cast(v16bf, fb[3 * KSTEPS * 32]);
      aL3 = WMMA_BF16(A0, Bm, aL3);  aH3 = WMMA_BF16(A1, Bm, aH3);
    }
    {
      v16bf Bm = __builtin_bit_cast(v16bf, fb[4 * KSTEPS * 32]);
      aL4 = WMMA_BF16(A0, Bm, aL4);  aH4 = WMMA_BF16(A1, Bm, aH4);
    }
    {
      v16bf Bm = __builtin_bit_cast(v16bf, fb[5 * KSTEPS * 32]);
      aL5 = WMMA_BF16(A0, Bm, aL5);  aH5 = WMMA_BF16(A1, Bm, aH5);
    }
    {
      v16bf Bm = __builtin_bit_cast(v16bf, fb[6 * KSTEPS * 32]);
      aL6 = WMMA_BF16(A0, Bm, aL6);  aH6 = WMMA_BF16(A1, Bm, aH6);
    }
  }

  // C layout: lane<16 -> rows M=j, lane>=16 -> rows M=8+j; col = nt*16 + l15
  {
    float* dL = ldsBig + (rowLocal + hsel * 8) * NCOLS + l15;        // rows 0..15 half
    float* dH = dL + 16 * NCOLS;                                     // rows 16..31 half
    #pragma unroll
    for (int j = 0; j < 8; ++j) {
      dL[j * NCOLS + 0 * 16] = aL0[j];  dH[j * NCOLS + 0 * 16] = aH0[j];
      dL[j * NCOLS + 1 * 16] = aL1[j];  dH[j * NCOLS + 1 * 16] = aH1[j];
      dL[j * NCOLS + 2 * 16] = aL2[j];  dH[j * NCOLS + 2 * 16] = aH2[j];
      dL[j * NCOLS + 3 * 16] = aL3[j];  dH[j * NCOLS + 3 * 16] = aH3[j];
      dL[j * NCOLS + 4 * 16] = aL4[j];  dH[j * NCOLS + 4 * 16] = aH4[j];
      dL[j * NCOLS + 5 * 16] = aL5[j];  dH[j * NCOLS + 5 * 16] = aH5[j];
      dL[j * NCOLS + 6 * 16] = aL6[j];  dH[j * NCOLS + 6 * 16] = aH6[j];
    }
  }

  // ---- small GEMM: prev_state (K=28, padded to 32) x [rkernel | Uo] ----
  {
    const float* hrow0 = prev_state + rowLo * V_DIM;
    const float* hrow1 = hrow0 + 16 * V_DIM;
    BFrag s0, s1;
    #pragma unroll
    for (int q = 0; q < 16; ++q) {
      int k = (q < 8) ? (o + q) : (16 + o + q - 8);
      s0.h[q] = (__bf16)((k < V_DIM) ? hrow0[k] : 0.f);
      s1.h[q] = (__bf16)((k < V_DIM) ? hrow1[k] : 0.f);
    }
    v16bf Ah0 = __builtin_bit_cast(v16bf, s0);
    v16bf Ah1 = __builtin_bit_cast(v16bf, s1);
    float* dL = ldsSmall + (rowLocal + hsel * 8) * NCOLS + l15;
    float* dH = dL + 16 * NCOLS;
    #pragma unroll
    for (int nt = 0; nt < NTILES; ++nt) {
      v16bf Bm = __builtin_bit_cast(v16bf, smallFrags[nt * 32 + lane]);
      v8f cL = WMMA_BF16(Ah0, Bm, zero);
      v8f cH = WMMA_BF16(Ah1, Bm, zero);
      #pragma unroll
      for (int j = 0; j < 8; ++j) {
        dL[j * NCOLS + nt * 16] = cL[j];
        dH[j * NCOLS + nt * 16] = cH[j];
      }
    }
  }

  __syncthreads();

  // ---- per-row GRU gates + logits + softmax (one thread per row) ----
  {
    const int r = tid;
    const long R = (long)blockIdx.x * ROWS_PER_BLOCK + r;
    const float* big = ldsBig + r * NCOLS;
    const float* sml = ldsSmall + r * NCOLS;
    float logits[V_DIM];
    float gru[V_DIM];
    for (int v = 0; v < V_DIM; ++v) {
      float h  = prev_state[R * V_DIM + v];
      float xz = big[v]        + gbias[v];
      float xr = big[28 + v]   + gbias[28 + v];
      float xh = big[56 + v]   + gbias[56 + v];
      float hz = sml[v]        + gbias[84 + v];
      float hr = sml[28 + v]   + gbias[84 + 28 + v];
      float hh = sml[56 + v]   + gbias[84 + 56 + v];
      float z  = 1.f / (1.f + expf(-(xz + hz)));
      float rr = 1.f / (1.f + expf(-(xr + hr)));
      float hc = tanhf(xh + rr * hh);
      gru[v] = z * h + (1.f - z) * hc;
      int idx = prev_pred[R * V_DIM + v];
      logits[v] = table[idx] + sml[84 + v] + big[84 + v] + Bo[v];
    }
    float m = logits[0];
    for (int v = 1; v < V_DIM; ++v) m = fmaxf(m, logits[v]);
    float s = 0.f;
    for (int v = 0; v < V_DIM; ++v) { logits[v] = expf(logits[v] - m); s += logits[v]; }
    float inv = 1.f / s;
    float* pred = out + R * V_DIM;
    float* go   = out + (long)B_ROWS * V_DIM + R * V_DIM;
    for (int v = 0; v < V_DIM; ++v) { pred[v] = logits[v] * inv; go[v] = gru[v]; }
  }
}

extern "C" void kernel_launch(void* const* d_in, const int* in_sizes, int n_in,
                              void* d_out, int out_size, void* d_ws, size_t ws_size,
                              hipStream_t stream) {
  const float* inputs = (const float*)d_in[0];
  const int*   ppred  = (const int*)d_in[1];
  const float* pstate = (const float*)d_in[2];
  const float* gk     = (const float*)d_in[3];
  const float* grk    = (const float*)d_in[4];
  const float* gbias  = (const float*)d_in[5];
  const float* Wo     = (const float*)d_in[6];
  const float* Uo     = (const float*)d_in[7];
  const float* Co     = (const float*)d_in[8];
  const float* Bo     = (const float*)d_in[9];
  const float* emb    = (const float*)d_in[10];
  float* out = (float*)d_out;

  uint16_t* wbig   = (uint16_t*)d_ws;                       // 114688 B
  uint16_t* wsmall = wbig + NTILES * KSTEPS * 32 * 16;      // +7168 B
  float*    table  = (float*)(wsmall + NTILES * 32 * 16);   // +112 B

  cgru_prep<<<64, 256, 0, stream>>>(gk, grk, Wo, Uo, Co, emb, wbig, wsmall, table);
  cgru_fused<<<B_ROWS / ROWS_PER_BLOCK, THREADS, 0, stream>>>(
      inputs, ppred, pstate, gbias, Bo, wbig, wsmall, table, out);
}
